// IntermediateLayerDistillation_21251498181020
// MI455X (gfx1250) — 16-bit and 8-bit matrix operand order repaired and checked by reading, compiled, not run on hardware
//
#include <hip/hip_runtime.h>
#include <hip/hip_bf16.h>


typedef __attribute__((ext_vector_type(8)))  float   v8f;
typedef __attribute__((ext_vector_type(16))) __bf16  v16bf;
typedef __attribute__((ext_vector_type(8)))  __bf16  v8bf;
typedef __attribute__((ext_vector_type(16))) int     v16i;
typedef __attribute__((ext_vector_type(4)))  int     v4i;
typedef __attribute__((ext_vector_type(2)))  int     v2i;

#define B_   4
#define S_   2048
#define DT_  4096
#define DS_  2048
#define ROWS_ (B_ * S_)
#define INV_MEAN_NEG (-1.0f / ((float)B_ * (float)S_ * (float)S_))


__device__ inline void async_load_b128(const void* gaddr, const void* lds_ptr) {
    unsigned int loff = (unsigned int)(size_t)lds_ptr;
    asm volatile("global_load_async_to_lds_b128 %0, %1, off"
                 :: "v"(loff), "v"(gaddr) : "memory");
}

template <int N>
__device__ inline void wait_asynccnt() {
#if __has_builtin(__builtin_amdgcn_s_wait_asynccnt)
    __builtin_amdgcn_s_wait_asynccnt((short)N);
#else
    asm volatile("s_wait_asynccnt %0" :: "i"(N) : "memory");
#endif
}


__device__ inline unsigned char f32_to_e4m3_sw(float f) {
    unsigned int s = (f < 0.0f) ? 0x80u : 0u;
    float af = fabsf(f);
    if (af > 448.0f) af = 448.0f;
    if (af < 0.0009765625f) return (unsigned char)s;
    unsigned int bits = __float_as_uint(af);
    int e = (int)(bits >> 23) - 127;
    if (e < -6) {
        unsigned int m = (unsigned int)rintf(af * 512.0f);
        return (unsigned char)(s | m);
    }
    unsigned int lsb = (bits >> 20) & 1u;
    bits += 0x7FFFFu + lsb;
    e = (int)(bits >> 23) - 127;
    unsigned int m = (bits >> 20) & 7u;
    int ef = e + 7;
    if (ef > 15) { ef = 15; m = 6; }
    return (unsigned char)(s | ((unsigned int)ef << 3) | m);
}

__device__ inline unsigned int pack4_fp8(float a, float b, float c, float d) {
#if __has_builtin(__builtin_amdgcn_cvt_pk_fp8_f32)
    int p = 0;
    p = __builtin_amdgcn_cvt_pk_fp8_f32(a, b, p, false);
    p = __builtin_amdgcn_cvt_pk_fp8_f32(c, d, p, true);
    return (unsigned int)p;
#else
    return (unsigned int)f32_to_e4m3_sw(a)
         | ((unsigned int)f32_to_e4m3_sw(b) << 8)
         | ((unsigned int)f32_to_e4m3_sw(c) << 16)
         | ((unsigned int)f32_to_e4m3_sw(d) << 24);
#endif
}


__global__ void zero_out_kernel(float* out) {
    if (threadIdx.x == 0) out[0] = 0.0f;
}


__global__ void __launch_bounds__(256) gemm_proj_bf16(
        const float* __restrict__ A, const float* __restrict__ W,
        const float* __restrict__ bias, float* __restrict__ C) {
    __shared__ __bf16 sA[128][40];
    __shared__ __bf16 sBt[128][40];

    const int tid  = threadIdx.x;
    const int lane = tid & 31;
    const int wave = tid >> 5;
    const int wm   = wave >> 1;
    const int wn   = wave & 1;
    const int m0   = blockIdx.y * 128;
    const int n0   = blockIdx.x * 128;
    const int r16  = lane & 15;
    const int aklo = (lane < 16) ? 0 : 8;
    const int bkb  = (lane < 16) ? 0 : 8;

    v8f acc[2][4] = {};

    for (int k0 = 0; k0 < DS_; k0 += 32) {
        __syncthreads();
#pragma unroll
        for (int i = 0; i < 4; ++i) {
            int idx = tid + 256 * i;
            int r = idx >> 3, c4 = idx & 7;
            float4 v = *(const float4*)&A[(size_t)(m0 + r) * DS_ + k0 + c4 * 4];
            sA[r][c4 * 4 + 0] = (__bf16)v.x;
            sA[r][c4 * 4 + 1] = (__bf16)v.y;
            sA[r][c4 * 4 + 2] = (__bf16)v.z;
            sA[r][c4 * 4 + 3] = (__bf16)v.w;
        }
#pragma unroll
        for (int i = 0; i < 4; ++i) {
            int idx = tid + 256 * i;
            int kr = idx >> 5, c4 = idx & 31;
            float4 v = *(const float4*)&W[(size_t)(k0 + kr) * DT_ + n0 + c4 * 4];
            sBt[c4 * 4 + 0][kr] = (__bf16)v.x;
            sBt[c4 * 4 + 1][kr] = (__bf16)v.y;
            sBt[c4 * 4 + 2][kr] = (__bf16)v.z;
            sBt[c4 * 4 + 3][kr] = (__bf16)v.w;
        }
        if (k0 + 32 < DS_) {
            __builtin_prefetch(&A[(size_t)(m0 + (tid >> 3)) * DS_ + k0 + 32], 0, 1);
            __builtin_prefetch(&W[(size_t)(k0 + 32 + (tid >> 5)) * DT_ + n0], 0, 1);
        }
        __syncthreads();

        v16bf afrag[2];
#pragma unroll
        for (int mt = 0; mt < 2; ++mt) {
            int r = wm * 32 + mt * 16 + r16;
            v8bf lo = *(const v8bf*)&sA[r][aklo];
            v8bf hi = *(const v8bf*)&sA[r][16 + aklo];
#pragma unroll
            for (int j = 0; j < 8; ++j) { afrag[mt][j] = lo[j]; afrag[mt][j + 8] = hi[j]; }
        }
#pragma unroll
        for (int nt = 0; nt < 4; ++nt) {
            int cn = wn * 64 + nt * 16 + r16;
            v8bf lo = *(const v8bf*)&sBt[cn][bkb];
            v8bf hi = *(const v8bf*)&sBt[cn][bkb + 16];
            v16bf bfrag;
#pragma unroll
            for (int j = 0; j < 8; ++j) { bfrag[j] = lo[j]; bfrag[j + 8] = hi[j]; }
#pragma unroll
            for (int mt = 0; mt < 2; ++mt) {
                acc[mt][nt] = __builtin_amdgcn_wmma_f32_16x16x32_bf16(
                    false, afrag[mt], false, bfrag, (short)0, acc[mt][nt], false, false);
            }
        }
    }

#pragma unroll
    for (int mt = 0; mt < 2; ++mt) {
#pragma unroll
        for (int nt = 0; nt < 4; ++nt) {
            int n = n0 + wn * 64 + nt * 16 + r16;
            float bv = bias[n];
#pragma unroll
            for (int r = 0; r < 8; ++r) {
                int m = m0 + wm * 32 + mt * 16 + r + ((lane >= 16) ? 8 : 0);
                C[(size_t)m * DT_ + n] = acc[mt][nt][r] + bv;
            }
        }
    }
}


__global__ void __launch_bounds__(256) rmsnorm_quant(
        const float* __restrict__ X, const float* __restrict__ gamma,
        unsigned char* __restrict__ Q, float* __restrict__ scl) {
    const int row = blockIdx.x;
    const int tid = threadIdx.x;
    const float* x = X + (size_t)row * DT_;

    float4 v[4];
    float ssq = 0.0f;
#pragma unroll
    for (int i = 0; i < 4; ++i) {
        v[i] = *(const float4*)&x[(tid + 256 * i) * 4];
        ssq += v[i].x * v[i].x + v[i].y * v[i].y + v[i].z * v[i].z + v[i].w * v[i].w;
    }

    __shared__ float sred[256];
    sred[tid] = ssq;
    __syncthreads();
    for (int s = 128; s > 0; s >>= 1) {
        if (tid < s) sred[tid] += sred[tid + s];
        __syncthreads();
    }
    const float rnorm = rsqrtf(sred[0] * (1.0f / (float)DT_) + 1e-6f);
    __syncthreads();

    float nv[16];
    float amax = 0.0f;
#pragma unroll
    for (int i = 0; i < 4; ++i) {
        float4 g = *(const float4*)&gamma[(tid + 256 * i) * 4];
        nv[4 * i + 0] = v[i].x * rnorm * g.x;
        nv[4 * i + 1] = v[i].y * rnorm * g.y;
        nv[4 * i + 2] = v[i].z * rnorm * g.z;
        nv[4 * i + 3] = v[i].w * rnorm * g.w;
        amax = fmaxf(amax, fabsf(nv[4 * i + 0]));
        amax = fmaxf(amax, fabsf(nv[4 * i + 1]));
        amax = fmaxf(amax, fabsf(nv[4 * i + 2]));
        amax = fmaxf(amax, fabsf(nv[4 * i + 3]));
    }
    sred[tid] = amax;
    __syncthreads();
    for (int s = 128; s > 0; s >>= 1) {
        if (tid < s) sred[tid] = fmaxf(sred[tid], sred[tid + s]);
        __syncthreads();
    }
    const float scale = fmaxf(sred[0] * (1.0f / 448.0f), 1e-6f);
    const float inv   = 1.0f / scale;
    if (tid == 0) scl[row] = scale;

    unsigned int* qo = (unsigned int*)(Q + (size_t)row * DT_);
#pragma unroll
    for (int i = 0; i < 4; ++i) {
        qo[tid + 256 * i] = pack4_fp8(nv[4 * i + 0] * inv, nv[4 * i + 1] * inv,
                                      nv[4 * i + 2] * inv, nv[4 * i + 3] * inv);
    }
}


__global__ void __launch_bounds__(256) sim_fp8_reduce(
        const unsigned char* __restrict__ Sq, const float* __restrict__ Ssc,
        const unsigned char* __restrict__ Tq, const float* __restrict__ Tsc,
        float* __restrict__ out) {
    __shared__ unsigned char sA[2][128][144];
    __shared__ unsigned char sB[2][128][144];
    __shared__ float s_ss[128], s_ts[128];
    __shared__ float sred[256];

    const int b   = blockIdx.z;
    const int tid = threadIdx.x;
    const int lane = tid & 31;
    const int wave = tid >> 5;
    const int wm = wave >> 1, wn = wave & 1;
    const int r16 = lane & 15;
    const int b8  = (lane < 16) ? 0 : 8;
    const int kb  = (lane < 16) ? 0 : 16;

    const unsigned char* Aq = Sq + (size_t)b * S_ * DT_ + (size_t)blockIdx.y * 128 * DT_;
    const unsigned char* Bq = Tq + (size_t)b * S_ * DT_ + (size_t)blockIdx.x * 128 * DT_;

    if (tid < 128) {
        s_ss[tid] = Ssc[b * S_ + blockIdx.y * 128 + tid];
        s_ts[tid] = Tsc[b * S_ + blockIdx.x * 128 + tid];
    }

    int srow[4], scol[4];
#pragma unroll
    for (int i = 0; i < 4; ++i) {
        int idx = tid + 256 * i;
        srow[i] = idx >> 3;
        scol[i] = (idx & 7) * 16;
    }

    auto issue_tile = [&](int buf, int k0) {
#pragma unroll
        for (int i = 0; i < 4; ++i) {
            async_load_b128(&Aq[(size_t)srow[i] * DT_ + k0 + scol[i]],
                            &sA[buf][srow[i]][scol[i]]);
            async_load_b128(&Bq[(size_t)srow[i] * DT_ + k0 + scol[i]],
                            &sB[buf][srow[i]][scol[i]]);
        }
    };

    v8f acc[2][4] = {};
    constexpr int NK = DT_ / 128;

    issue_tile(0, 0);

    for (int kt = 0; kt < NK; ++kt) {
        const int p = kt & 1;
        __syncthreads();
        if (kt + 1 < NK) {
            issue_tile(1 - p, (kt + 1) * 128);
            wait_asynccnt<8>();
        } else {
            wait_asynccnt<0>();
        }
        __syncthreads();

        v16i afrag[2];
#pragma unroll
        for (int mt = 0; mt < 2; ++mt) {
            int row = wm * 32 + mt * 16 + r16;
#pragma unroll
            for (int j = 0; j < 4; ++j) {
                v4i t = *(const v4i*)&sA[p][row][j * 32 + kb];
                afrag[mt][4 * j + 0] = t[0];
                afrag[mt][4 * j + 1] = t[1];
                afrag[mt][4 * j + 2] = t[2];
                afrag[mt][4 * j + 3] = t[3];
            }
        }
#pragma unroll
        for (int nt = 0; nt < 4; ++nt) {
            int col = wn * 64 + nt * 16 + r16;
            v16i bfrag;
#pragma unroll
            for (int g = 0; g < 4; ++g) {
                v4i t = *(const v4i*)&sB[p][col][g * 32 + kb];
                bfrag[4 * g + 0] = t[0];
                bfrag[4 * g + 1] = t[1];
                bfrag[4 * g + 2] = t[2];
                bfrag[4 * g + 3] = t[3];
            }
#pragma unroll
            for (int mt = 0; mt < 2; ++mt) {
                acc[mt][nt] = __builtin_amdgcn_wmma_f32_16x16x128_fp8_fp8(
                    afrag[mt], bfrag, (short)0, acc[mt][nt], false, false);
            }
        }
    }

    float part = 0.0f;
#pragma unroll
    for (int mt = 0; mt < 2; ++mt) {
#pragma unroll
        for (int nt = 0; nt < 4; ++nt) {
            int nl = wn * 64 + nt * 16 + r16;
            float tsv = s_ts[nl];
#pragma unroll
            for (int r = 0; r < 8; ++r) {
                int ml = wm * 32 + mt * 16 + r + ((lane >= 16) ? 8 : 0);
                part += acc[mt][nt][r] * s_ss[ml] * tsv;
            }
        }
    }
    sred[tid] = part;
    __syncthreads();
    for (int s = 128; s > 0; s >>= 1) {
        if (tid < s) sred[tid] += sred[tid + s];
        __syncthreads();
    }
    if (tid == 0) atomicAdd(out, sred[0] * INV_MEAN_NEG);
}


extern "C" void kernel_launch(void* const* d_in, const int* in_sizes, int n_in,
                              void* d_out, int out_size, void* d_ws, size_t ws_size,
                              hipStream_t stream) {
    const float* teacher = (const float*)d_in[0];
    const float* student = (const float*)d_in[1];
    const float* w_proj  = (const float*)d_in[2];
    const float* b_proj  = (const float*)d_in[3];
    const float* t_gamma = (const float*)d_in[4];
    const float* s_gamma = (const float*)d_in[5];
    float* out = (float*)d_out;

    float* proj = (float*)d_ws;
    unsigned char* tq = (unsigned char*)(proj + (size_t)ROWS_ * DT_);
    unsigned char* sq = tq + (size_t)ROWS_ * DT_;
    float* ts = (float*)(sq + (size_t)ROWS_ * DT_);
    float* ss = ts + ROWS_;

    zero_out_kernel<<<1, 32, 0, stream>>>(out);

    gemm_proj_bf16<<<dim3(DT_ / 128, ROWS_ / 128), 256, 0, stream>>>(
        student, w_proj, b_proj, proj);

    rmsnorm_quant<<<ROWS_, 256, 0, stream>>>(teacher, t_gamma, tq, ts);
    rmsnorm_quant<<<ROWS_, 256, 0, stream>>>(proj, s_gamma, sq, ss);

    sim_fp8_reduce<<<dim3(S_ / 128, S_ / 128, B_), 256, 0, stream>>>(
        sq, ss, tq, ts, out);
}
